// _LocalTensorProductLayer_34849364640433
// MI455X (gfx1250) — matrix operand order repaired and checked by reading, not compiled, not run on hardware
//
#include <hip/hip_runtime.h>
#include <hip/hip_bf16.h>
#include <stdint.h>

typedef __bf16 bf16;
typedef __attribute__((ext_vector_type(16))) __bf16 v16bf;
typedef __attribute__((ext_vector_type(8)))  __bf16 v8bf;
typedef __attribute__((ext_vector_type(8)))  float  v8f;

#define WMMA_BF16(a, b, c) \
  __builtin_amdgcn_wmma_f32_16x16x32_bf16(false, (a), false, (b), (short)0, (c), false, false)

// ---------------------------------------------------------------------------
// A-operand loaders.  CDNA5 16-bit A 16x32 layout (wave32):
//   lanes 0-15  : M = lane,    K in {0..7, 16..23}
//   lanes 16-31 : M = lane-16, K in {8..15, 24..31}
// so each lane needs two contiguous 8-element (16B) runs.
// ---------------------------------------------------------------------------
__device__ inline v16bf load_a_bf16(const bf16* __restrict__ p, int hi) {
  union { v16bf v; v8bf h[2]; } u;
  u.h[0] = *(const v8bf*)(p + hi * 8);
  u.h[1] = *(const v8bf*)(p + 16 + hi * 8);
  return u.v;
}

__device__ inline v16bf cvt_a_f32(const float* __restrict__ p, int hi) {
  v16bf a;
#pragma unroll
  for (int t = 0; t < 8; ++t) a[t] = (bf16)p[hi * 8 + t];
#pragma unroll
  for (int t = 0; t < 8; ++t) a[8 + t] = (bf16)p[16 + hi * 8 + t];
  return a;
}

__device__ inline float silu(float x) { return x / (1.0f + __expf(-x)); }

// ---------------------------------------------------------------------------
// Prep kernels (run once per launch; all tiny vs. the edge stage)
// ---------------------------------------------------------------------------

// f32 node features -> bf16 copy
__global__ void cvt_nodes_kernel(const float* __restrict__ src,
                                 bf16* __restrict__ dst, int n) {
  int i = blockIdx.x * 256 + threadIdx.x;
  if (i < n) dst[i] = (bf16)src[i];
}

// Pack a row-major f32 weight [K,N] into the WMMA-B wave32 layout:
//  dstidx = ((nt*nChunks + chunk)*32 + lane)*16 + t
//  k = chunk*32 + (t<8 ? (lane>>4)*8 + t : 16 + (lane>>4)*8 + (t-8)) ; n = nt*16 + (lane&15)
__global__ void pack_b_kernel(bf16* __restrict__ dst, const float* __restrict__ W,
                              int N, int Kvalid, int nChunks, int total) {
  int idx = blockIdx.x * 256 + threadIdx.x;
  if (idx >= total) return;
  int t    = idx & 15;
  int lane = (idx >> 4) & 31;
  int nc   = idx >> 9;                 // nt*nChunks + chunk
  int chunk = nc % nChunks;
  int nt    = nc / nChunks;
  int k = chunk * 32 + ((t < 8) ? ((lane >> 4) * 8 + t) : (16 + (lane >> 4) * 8 + (t - 8)));
  int n = nt * 16 + (lane & 15);
  dst[idx] = (k < Kvalid) ? (bf16)W[(size_t)k * N + n] : (bf16)0.0f;
}

// Composite weight Wc = Wa(128x128) @ Wf1[rowOff:rowOff+128, 0:128], packed
// directly into WMMA-B layout (8 n-tiles x 4 k-chunks).
__global__ void pack_composite_kernel(bf16* __restrict__ dst,
                                      const float* __restrict__ Wa,
                                      const float* __restrict__ Wf1, int rowOff) {
  int idx = blockIdx.x * 256 + threadIdx.x;   // total = 8*4*32*16 = 16384
  if (idx >= 16384) return;
  int t    = idx & 15;
  int lane = (idx >> 4) & 31;
  int nc   = idx >> 9;
  int chunk = nc & 3;
  int nt    = nc >> 2;
  int k = chunk * 32 + ((t < 8) ? ((lane >> 4) * 8 + t) : (16 + (lane >> 4) * 8 + (t - 8)));
  int n = nt * 16 + (lane & 15);
  float s = 0.0f;
  for (int j = 0; j < 128; ++j)
    s += Wa[k * 128 + j] * Wf1[(size_t)(rowOff + j) * 128 + n];
  dst[idx] = (bf16)s;
}

// bc[n] = b_f1[n] + b_src @ Wf1[0:128] + b_dst @ Wf1[128:256]
__global__ void bias_compose_kernel(float* __restrict__ bc,
                                    const float* __restrict__ bsrc,
                                    const float* __restrict__ bdst,
                                    const float* __restrict__ Wf1,
                                    const float* __restrict__ bf1) {
  int n = threadIdx.x;   // 128 threads
  float s = bf1[n];
  for (int j = 0; j < 128; ++j)
    s += bsrc[j] * Wf1[j * 128 + n] + bdst[j] * Wf1[(128 + j) * 128 + n];
  bc[n] = s;
}

// ---------------------------------------------------------------------------
// Edge stage: per 16-edge tile (one wave):
//   pre = xb[src]@WcS + xb[dst]@WcD + rbf@Wr + bc   (WMMA, bf16)
//   msg = silu(pre) @ Wf2 + b_f2                    (WMMA via LDS transpose)
//   atomics: agg += msg/12 ; T1 += msg*basis1 ; T2 += msg*basis2
// ---------------------------------------------------------------------------
__global__ __launch_bounds__(128)
void edge_kernel(const bf16* __restrict__ xb,
                 const long long* __restrict__ ei,      // [2,E]
                 const float* __restrict__ rbf,         // [E,16]
                 const float* __restrict__ basis1,      // [E,3]
                 const float* __restrict__ basis2,      // [E,5]
                 const bf16* __restrict__ pWcS,         // 8nt x 4ch
                 const bf16* __restrict__ pWcD,         // 8nt x 4ch
                 const bf16* __restrict__ pWr,          // 8nt x 1ch
                 const bf16* __restrict__ pWf2,         // 8nt x 4ch
                 const float* __restrict__ bc,          // [128]
                 const float* __restrict__ bf2,         // [128]
                 float* __restrict__ agg,               // [N,128]
                 float* __restrict__ T1,                // [N,128,3]
                 float* __restrict__ T2,                // [N,128,5]
                 int E, int nTiles) {
  __shared__ bf16 sMsg[4][16 * 128];
  const int wave = threadIdx.x >> 5;
  const int lane = threadIdx.x & 31;
  const int tile = blockIdx.x * 4 + wave;
  if (tile >= nTiles) return;

  const int e0  = tile * 16;
  const int row = lane & 15;
  const int hi  = lane >> 4;
  const int e   = e0 + row;

  const long long sIdx = ei[e];             // src node for this lane's row
  const long long dIdx = ei[(size_t)E + e]; // dst node for this lane's row

  // ---- A fragments for GEMM1 -------------------------------------------
  v16bf aS[4], aD[4];
#pragma unroll
  for (int c = 0; c < 4; ++c) {
    aS[c] = load_a_bf16(xb + (size_t)sIdx * 128 + c * 32, hi);
    aD[c] = load_a_bf16(xb + (size_t)dIdx * 128 + c * 32, hi);
  }
  // rbf chunk: only K=0..15 valid -> elements 0..7 hold K = hi*8..hi*8+7
  v16bf aR;
  {
    const float* rp = rbf + (size_t)e * 16;
#pragma unroll
    for (int t = 0; t < 8; ++t) aR[t] = (bf16)rp[hi * 8 + t];
#pragma unroll
    for (int t = 0; t < 8; ++t) aR[8 + t] = (bf16)0.0f;
  }

  // ---- GEMM1: pre = A @ [WcS|WcD|Wr] + bc ; silu ; transpose via LDS ----
  bf16* myLds = sMsg[wave];
  for (int nt = 0; nt < 8; ++nt) {
    v8f acc = {};
#pragma unroll
    for (int c = 0; c < 4; ++c) {
      v16bf bS = *(const v16bf*)(pWcS + ((size_t)(nt * 4 + c) * 32 + lane) * 16);
      acc = WMMA_BF16(aS[c], bS, acc);
      v16bf bD = *(const v16bf*)(pWcD + ((size_t)(nt * 4 + c) * 32 + lane) * 16);
      acc = WMMA_BF16(aD[c], bD, acc);
    }
    v16bf bR = *(const v16bf*)(pWr + ((size_t)nt * 32 + lane) * 16);
    acc = WMMA_BF16(aR, bR, acc);

    const float bb = bc[nt * 16 + row];
#pragma unroll
    for (int j = 0; j < 8; ++j) {
      const int m = j + hi * 8;              // C/D layout: row m, col nt*16+row
      myLds[m * 128 + nt * 16 + row] = (bf16)silu(acc[j] + bb);
    }
  }

  // ---- GEMM2: msg = silu(pre) @ Wf2 ; fused atomic scatter --------------
  v16bf a2[4];
#pragma unroll
  for (int c = 0; c < 4; ++c)
    a2[c] = load_a_bf16(myLds + row * 128 + c * 32, hi);

  const float inv12 = 1.0f / 12.0f;
  for (int nt = 0; nt < 8; ++nt) {
    v8f acc = {};
#pragma unroll
    for (int c = 0; c < 4; ++c) {
      v16bf b = *(const v16bf*)(pWf2 + ((size_t)(nt * 4 + c) * 32 + lane) * 16);
      acc = WMMA_BF16(a2[c], b, acc);
    }
    const int   h    = nt * 16 + row;
    const float bias = bf2[h];
#pragma unroll
    for (int j = 0; j < 8; ++j) {
      const int m  = j + hi * 8;
      const int em = e0 + m;
      const long long dm = ei[(size_t)E + em];
      const float v = acc[j] + bias;
      const size_t base = (size_t)dm * 128 + h;
      atomicAdd(agg + base, v * inv12);
      const float* b1 = basis1 + (size_t)em * 3;
      atomicAdd(T1 + base * 3 + 0, v * b1[0]);
      atomicAdd(T1 + base * 3 + 1, v * b1[1]);
      atomicAdd(T1 + base * 3 + 2, v * b1[2]);
      const float* b2 = basis2 + (size_t)em * 5;
      atomicAdd(T2 + base * 5 + 0, v * b2[0]);
      atomicAdd(T2 + base * 5 + 1, v * b2[1]);
      atomicAdd(T2 + base * 5 + 2, v * b2[2]);
      atomicAdd(T2 + base * 5 + 3, v * b2[3]);
      atomicAdd(T2 + base * 5 + 4, v * b2[4]);
    }
  }
}

// ---------------------------------------------------------------------------
// Node stage: per 16-node tile (one wave):
//   ui  = [x | agg | mean(T1^2) | mean(T2^2)]   (512-dim, built as A frags)
//   u   = silu(ui @ Wu1 + b_u1) @ Wu2 + b_u2    (WMMA x2 via LDS transpose)
//   out = LayerNorm(x + u) * g + b
// ---------------------------------------------------------------------------
__global__ __launch_bounds__(128)
void node_kernel(const float* __restrict__ nf,
                 const bf16* __restrict__ xb,
                 const float* __restrict__ agg,
                 const float* __restrict__ T1,
                 const float* __restrict__ T2,
                 const bf16* __restrict__ pWu1,  // 16nt x 16ch
                 const float* __restrict__ bu1,  // [256]
                 const bf16* __restrict__ pWu2,  // 8nt x 8ch
                 const float* __restrict__ bu2,  // [128]
                 const float* __restrict__ lng,
                 const float* __restrict__ lnb,
                 float* __restrict__ out,
                 int nTiles) {
  __shared__ bf16  sU[4][16 * 256];
  __shared__ float sR[4][16 * 128];
  const int wave = threadIdx.x >> 5;
  const int lane = threadIdx.x & 31;
  const int tile = blockIdx.x * 4 + wave;
  if (tile >= nTiles) return;

  const int n0   = tile * 16;
  const int row  = lane & 15;
  const int hi   = lane >> 4;
  const size_t nrow = (size_t)(n0 + row);

  // ---- build 16 K-chunks of the 512-wide update input ------------------
  v16bf A[16];
#pragma unroll
  for (int c = 0; c < 4; ++c)
    A[c] = load_a_bf16(xb + nrow * 128 + c * 32, hi);
#pragma unroll
  for (int c = 0; c < 4; ++c)
    A[4 + c] = cvt_a_f32(agg + nrow * 128 + c * 32, hi);
#pragma unroll
  for (int c = 0; c < 4; ++c) {
    v16bf a1, a2v;
#pragma unroll
    for (int t = 0; t < 16; ++t) {
      const int kk = (t < 8) ? (hi * 8 + t) : (16 + hi * 8 + (t - 8));
      const size_t hidx = nrow * 128 + c * 32 + kk;
      const float* p1 = T1 + hidx * 3;
      a1[t] = (bf16)((p1[0] * p1[0] + p1[1] * p1[1] + p1[2] * p1[2]) * (1.0f / 3.0f));
      const float* p2 = T2 + hidx * 5;
      a2v[t] = (bf16)((p2[0] * p2[0] + p2[1] * p2[1] + p2[2] * p2[2] +
                       p2[3] * p2[3] + p2[4] * p2[4]) * (1.0f / 5.0f));
    }
    A[8 + c]  = a1;
    A[12 + c] = a2v;
  }

  // ---- GEMM u1 (512 -> 256), silu, LDS transpose ------------------------
  bf16* myU = sU[wave];
  for (int nt = 0; nt < 16; ++nt) {
    v8f acc = {};
#pragma unroll
    for (int c = 0; c < 16; ++c) {
      v16bf b = *(const v16bf*)(pWu1 + ((size_t)(nt * 16 + c) * 32 + lane) * 16);
      acc = WMMA_BF16(A[c], b, acc);
    }
    const float bb = bu1[nt * 16 + row];
#pragma unroll
    for (int j = 0; j < 8; ++j)
      myU[(j + hi * 8) * 256 + nt * 16 + row] = (bf16)silu(acc[j] + bb);
  }

  // ---- GEMM u2 (256 -> 128), residual into LDS --------------------------
  v16bf A2[8];
#pragma unroll
  for (int c = 0; c < 8; ++c)
    A2[c] = load_a_bf16(myU + row * 256 + c * 32, hi);

  float* myR = sR[wave];
  for (int nt = 0; nt < 8; ++nt) {
    v8f acc = {};
#pragma unroll
    for (int c = 0; c < 8; ++c) {
      v16bf b = *(const v16bf*)(pWu2 + ((size_t)(nt * 8 + c) * 32 + lane) * 16);
      acc = WMMA_BF16(A2[c], b, acc);
    }
    const int   n    = nt * 16 + row;
    const float bias = bu2[n];
#pragma unroll
    for (int j = 0; j < 8; ++j) {
      const int m = j + hi * 8;
      myR[m * 128 + n] = nf[(size_t)(n0 + m) * 128 + n] + acc[j] + bias;
    }
  }

  // ---- LayerNorm: lanes (l, l^16) cooperate on row l&15 ------------------
  const float* R = myR + row * 128;
  float s1 = 0.0f, s2 = 0.0f;
  for (int i = hi * 64; i < hi * 64 + 64; ++i) {
    const float x = R[i];
    s1 += x; s2 += x * x;
  }
  s1 += __shfl_xor(s1, 16);
  s2 += __shfl_xor(s2, 16);
  const float mu   = s1 * (1.0f / 128.0f);
  const float var  = s2 * (1.0f / 128.0f) - mu * mu;
  const float rstd = rsqrtf(var + 1e-5f);
  float* op = out + nrow * 128;
  for (int i = hi * 64; i < hi * 64 + 64; ++i)
    op[i] = (R[i] - mu) * rstd * lng[i] + lnb[i];
}

// ---------------------------------------------------------------------------
// Launch
// ---------------------------------------------------------------------------
extern "C" void kernel_launch(void* const* d_in, const int* in_sizes, int n_in,
                              void* d_out, int out_size, void* d_ws, size_t ws_size,
                              hipStream_t stream) {
  const float*     nf   = (const float*)d_in[0];
  const long long* ei   = (const long long*)d_in[1];
  const float*     rbf  = (const float*)d_in[2];
  // d_in[3] (basis_l0) intentionally unused by the reference
  const float*     bas1 = (const float*)d_in[4];
  const float*     bas2 = (const float*)d_in[5];
  const float*     Wsrc = (const float*)d_in[6];
  const float*     bsrc = (const float*)d_in[7];
  const float*     Wdst = (const float*)d_in[8];
  const float*     bdst = (const float*)d_in[9];
  const float*     Wf1  = (const float*)d_in[10];
  const float*     bf1  = (const float*)d_in[11];
  const float*     Wf2  = (const float*)d_in[12];
  const float*     bf2  = (const float*)d_in[13];
  const float*     Wu1  = (const float*)d_in[14];
  const float*     bu1  = (const float*)d_in[15];
  const float*     Wu2  = (const float*)d_in[16];
  const float*     bu2  = (const float*)d_in[17];
  const float*     lng  = (const float*)d_in[18];
  const float*     lnb  = (const float*)d_in[19];
  float*           outp = (float*)d_out;

  const int NN = in_sizes[0] / 128;   // 20000
  const int E  = in_sizes[2] / 16;    // 320000

  size_t off = 0;
  auto take = [&](size_t bytes) -> void* {
    void* p = (char*)d_ws + off;
    off += (bytes + 255) & ~(size_t)255;
    return p;
  };
  bf16*  xb   = (bf16*)take((size_t)NN * 128 * sizeof(bf16));
  bf16*  pWcS = (bf16*)take(16384 * sizeof(bf16));   // 8nt x 4ch
  bf16*  pWcD = (bf16*)take(16384 * sizeof(bf16));
  bf16*  pWr  = (bf16*)take(4096  * sizeof(bf16));   // 8nt x 1ch
  bf16*  pWf2 = (bf16*)take(16384 * sizeof(bf16));
  bf16*  pWu1 = (bf16*)take(131072 * sizeof(bf16));  // 16nt x 16ch
  bf16*  pWu2 = (bf16*)take(32768 * sizeof(bf16));   // 8nt x 8ch
  float* bc   = (float*)take(128 * sizeof(float));
  float* agg  = (float*)take((size_t)NN * 128 * sizeof(float));
  float* T1   = (float*)take((size_t)NN * 128 * 3 * sizeof(float));
  float* T2   = (float*)take((size_t)NN * 128 * 5 * sizeof(float));

  // zero the atomic accumulators
  hipMemsetAsync(agg, 0, (size_t)NN * 128 * sizeof(float), stream);
  hipMemsetAsync(T1,  0, (size_t)NN * 128 * 3 * sizeof(float), stream);
  hipMemsetAsync(T2,  0, (size_t)NN * 128 * 5 * sizeof(float), stream);

  // prep
  {
    const int n = NN * 128;
    cvt_nodes_kernel<<<(n + 255) / 256, 256, 0, stream>>>(nf, xb, n);
  }
  pack_composite_kernel<<<64, 256, 0, stream>>>(pWcS, Wsrc, Wf1, 0);
  pack_composite_kernel<<<64, 256, 0, stream>>>(pWcD, Wdst, Wf1, 128);
  pack_b_kernel<<<(4096 + 255) / 256, 256, 0, stream>>>(pWr,  Wf1 + 256 * 128, 128, 16,  1,  4096);
  pack_b_kernel<<<(16384 + 255) / 256, 256, 0, stream>>>(pWf2, Wf2, 128, 128, 4,  16384);
  pack_b_kernel<<<(131072 + 255) / 256, 256, 0, stream>>>(pWu1, Wu1, 256, 512, 16, 131072);
  pack_b_kernel<<<(32768 + 255) / 256, 256, 0, stream>>>(pWu2, Wu2, 128, 256, 8,  32768);
  bias_compose_kernel<<<1, 128, 0, stream>>>(bc, bsrc, bdst, Wf1, bf1);

  // edge stage: one 16-edge tile per wave, 4 waves/block
  {
    const int tiles = E / 16;
    edge_kernel<<<(tiles + 3) / 4, 128, 0, stream>>>(
        xb, ei, rbf, bas1, bas2, pWcS, pWcD, pWr, pWf2, bc, bf2,
        agg, T1, T2, E, tiles);
  }
  // node stage: one 16-node tile per wave, 4 waves/block
  {
    const int tiles = NN / 16;
    node_kernel<<<(tiles + 3) / 4, 128, 0, stream>>>(
        nf, xb, agg, T1, T2, pWu1, bu1, pWu2, bu2, lng, lnb, outp, tiles);
  }
}
